// DiffNetModel_53635551592548
// MI455X (gfx1250) — compile-verified, matrix operand order checked by reading, not run on hardware
//
#include <hip/hip_runtime.h>

typedef __attribute__((ext_vector_type(16))) _Float16 v16h;
typedef __attribute__((ext_vector_type(4)))  _Float16 v4h;
typedef __attribute__((ext_vector_type(8)))  float    v8f;

#define NF   64     // factors
#define NL   2      // layers
#define KN   50     // max neighbors
#define TILE 16     // batch rows per block

__global__ __launch_bounds__(128) void diffnet_fused_kernel(
    const int*   __restrict__ user_idx,
    const int*   __restrict__ product_idx,
    const int*   __restrict__ category_idx,
    const int*   __restrict__ neighbor_idx,
    const int*   __restrict__ neighbor_lens,
    const float* __restrict__ user_emb,
    const float* __restrict__ product_emb,
    const float* __restrict__ category_emb,
    const float* __restrict__ user_bias,
    const float* __restrict__ product_bias,
    const float* __restrict__ global_bias,
    const float* __restrict__ Wmat,    // [L,F,F]  (row n holds W[l][n][k])
    const float* __restrict__ bvec,    // [L,F]
    float*       __restrict__ out,     // [B]
    int B)
{
    __shared__ float    lds_u[TILE][NF];      // current user vector u_l
    __shared__ float    lds_mean[TILE][NF];   // neighbor mean (fixed)
    __shared__ _Float16 lds_x[TILE][NF];      // x = u + mean, f16 (GEMM A input)
    __shared__ __align__(16) _Float16 lds_w[NL][NF][NF];  // W in f16
    __shared__ float    lds_hn[TILE];         // has-neighbor mask as 0.0/1.0

    const int tid  = threadIdx.x;
    const int wave = tid >> 5;
    const int lane = tid & 31;
    const int row0 = blockIdx.x * TILE;

    // ---- stage W -> LDS as f16, vectorized (float4 in, 8B ds_store out) ----
    for (int i = tid; i < NL * NF * NF / 4; i += 128) {
        const float4 w4 = ((const float4*)Wmat)[i];
        v4h h;
        h[0] = (_Float16)w4.x; h[1] = (_Float16)w4.y;
        h[2] = (_Float16)w4.z; h[3] = (_Float16)w4.w;
        ((v4h*)lds_w)[i] = h;
    }

    // ---- phase 1: gather u rows + masked neighbor mean ----
    // wave w handles rows w*4 .. w*4+3; each lane owns 2 features (float2)
    for (int r = 0; r < 4; ++r) {
        const int m   = wave * 4 + r;
        const int row = row0 + m;
        if (row < B) {
            const int uidx = user_idx[row];
            const float2 uu = *(const float2*)(user_emb + (long)uidx * NF + lane * 2);

            int len = neighbor_lens[row];
            len = (len > KN) ? KN : (len < 0 ? 0 : len);

            // cooperative index fetch: lanes 0..31 -> k, lanes 0..17 -> k+32
            const long kbase = (long)row * KN;
            const int idx_a = (lane < len)      ? neighbor_idx[kbase + lane]      : 0;
            const int idx_b = (lane + 32 < len) ? neighbor_idx[kbase + 32 + lane] : 0;

            float2 acc0 = make_float2(0.f, 0.f);
            float2 acc1 = make_float2(0.f, 0.f);
            int k = 0;
            for (; k + 2 <= len; k += 2) {
                // k is wave-uniform -> readlane gives an SGPR index -> saddr loads
                const int n0 = (k < 32) ? __builtin_amdgcn_readlane(idx_a, k)
                                        : __builtin_amdgcn_readlane(idx_b, k - 32);
                const int k1 = k + 1;
                const int n1 = (k1 < 32) ? __builtin_amdgcn_readlane(idx_a, k1)
                                         : __builtin_amdgcn_readlane(idx_b, k1 - 32);
                const float2 v0 = *(const float2*)(user_emb + (long)n0 * NF + lane * 2);
                const float2 v1 = *(const float2*)(user_emb + (long)n1 * NF + lane * 2);
                acc0.x += v0.x; acc0.y += v0.y;
                acc1.x += v1.x; acc1.y += v1.y;
            }
            if (k < len) {
                const int n0 = (k < 32) ? __builtin_amdgcn_readlane(idx_a, k)
                                        : __builtin_amdgcn_readlane(idx_b, k - 32);
                const float2 v0 = *(const float2*)(user_emb + (long)n0 * NF + lane * 2);
                acc0.x += v0.x; acc0.y += v0.y;
            }
            const float inv = 1.0f / fmaxf((float)len, 1.0f);
            const float mx = (acc0.x + acc1.x) * inv;
            const float my = (acc0.y + acc1.y) * inv;

            lds_u[m][lane * 2 + 0]    = uu.x;
            lds_u[m][lane * 2 + 1]    = uu.y;
            lds_mean[m][lane * 2 + 0] = mx;
            lds_mean[m][lane * 2 + 1] = my;
            lds_x[m][lane * 2 + 0]    = (_Float16)(uu.x + mx);
            lds_x[m][lane * 2 + 1]    = (_Float16)(uu.y + my);
            if (lane == 0) lds_hn[m] = (len > 0) ? 1.0f : 0.0f;
        }
    }
    __syncthreads();

    // ---- phase 2: L diffusion layers via WMMA ----
    // Each wave computes output columns n = wave*16 .. wave*16+15 for all 16 rows.
    const int ncol = (lane & 15) + wave * 16;  // output column (N) for this lane
    const int koff = (lane >> 4) << 3;         // A/B K-offset for upper lane half
    const int moff = koff;                     // C/D M-offset for upper lane half

    for (int l = 0; l < NL; ++l) {
        v16h a0, a1, b0, b1;
        #pragma unroll
        for (int e = 0; e < 16; ++e) {
            const int kk = ((e >> 3) << 4) + koff + (e & 7);
            a0[e] = lds_x[lane & 15][kk];
            a1[e] = lds_x[lane & 15][kk + 32];
            b0[e] = lds_w[l][ncol][kk];
            b1[e] = lds_w[l][ncol][kk + 32];
        }
        v8f c;
        const float bias = bvec[l * NF + ncol];
        #pragma unroll
        for (int r = 0; r < 8; ++r) c[r] = bias;

        c = __builtin_amdgcn_wmma_f32_16x16x32_f16(false, a0, false, b0,
                                                   (short)0, c, false, false);
        c = __builtin_amdgcn_wmma_f32_16x16x32_f16(false, a1, false, b1,
                                                   (short)0, c, false, false);

        // branchless relu + empty-neighbor passthrough:
        //   c = prev + hn * (relu(c) - prev)
        #pragma unroll
        for (int r = 0; r < 8; ++r) {
            const int m = r + moff;
            const float prev = lds_u[m][ncol];
            const float hn   = lds_hn[m];
            const float rl   = fmaxf(c[r], 0.0f);
            c[r] = __builtin_fmaf(hn, rl - prev, prev);
        }
        __syncthreads();   // everyone done reading lds_u / lds_x of this layer
        #pragma unroll
        for (int r = 0; r < 8; ++r) {
            const int m = r + moff;
            lds_u[m][ncol] = c[r];
            if (l + 1 < NL)
                lds_x[m][ncol] = (_Float16)(c[r] + lds_mean[m][ncol]);
        }
        __syncthreads();
    }

    // ---- phase 3: interaction + biases ----
    // 8 lanes per row; lane j sums features j, j+8, ..., j+56
    {
        const int m   = tid >> 3;
        const int j   = tid & 7;
        const int row = row0 + m;
        if (row < B) {
            const int pidx = product_idx[row];
            const int cidx = category_idx[row];
            float s = 0.f;
            #pragma unroll
            for (int f = j; f < NF; f += 8) {
                const float cp = product_emb[(long)pidx * NF + f]
                               + 0.3f * category_emb[(long)cidx * NF + f];
                s += lds_u[m][f] * cp;
            }
            s += __shfl_xor(s, 4, 8);
            s += __shfl_xor(s, 2, 8);
            s += __shfl_xor(s, 1, 8);
            if (j == 0) {
                out[row] = global_bias[0] + user_bias[user_idx[row]]
                         + product_bias[pidx] + s;
            }
        }
    }
}

extern "C" void kernel_launch(void* const* d_in, const int* in_sizes, int n_in,
                              void* d_out, int out_size, void* d_ws, size_t ws_size,
                              hipStream_t stream) {
    const int*   user_idx      = (const int*)d_in[0];
    const int*   product_idx   = (const int*)d_in[1];
    const int*   category_idx  = (const int*)d_in[2];
    const int*   neighbor_idx  = (const int*)d_in[3];
    const int*   neighbor_lens = (const int*)d_in[4];
    const float* user_emb      = (const float*)d_in[5];
    const float* product_emb   = (const float*)d_in[6];
    const float* category_emb  = (const float*)d_in[7];
    const float* user_bias     = (const float*)d_in[8];
    const float* product_bias  = (const float*)d_in[9];
    const float* global_bias   = (const float*)d_in[10];
    const float* Wmat          = (const float*)d_in[11];
    const float* bvec          = (const float*)d_in[12];
    float*       out           = (float*)d_out;

    const int B = in_sizes[0];                 // 16384
    const int grid = (B + TILE - 1) / TILE;    // 1024 blocks of 16 rows

    diffnet_fused_kernel<<<grid, 128, 0, stream>>>(
        user_idx, product_idx, category_idx, neighbor_idx, neighbor_lens,
        user_emb, product_emb, category_emb, user_bias, product_bias,
        global_bias, Wmat, bvec, out, B);
}
